// GraphAttentionLayer_16698832847056
// MI455X (gfx1250) — matrix operand order repaired and checked by reading, not compiled, not run on hardware
//
#include <hip/hip_runtime.h>
#include <hip/hip_bf16.h>

#define IN_F   128
#define OUT_F  64
#define ALPHA  0.2f
#define EPSV   1e-8f

typedef __attribute__((ext_vector_type(16))) __bf16 v16bf;
typedef __attribute__((ext_vector_type(2)))  __bf16 v2bf;
typedef __attribute__((ext_vector_type(8)))  float  v8f;

// Monotone order-preserving float<->uint mapping for atomicMax-based segment max.
__device__ __forceinline__ unsigned enc_f32(float v) {
    unsigned b = __float_as_uint(v);
    return (b & 0x80000000u) ? ~b : (b | 0x80000000u);
}
__device__ __forceinline__ float dec_f32(unsigned e) {
    unsigned b = (e & 0x80000000u) ? (e & 0x7FFFFFFFu) : ~e;
    return __uint_as_float(b);
}
#define ENC_NEG_INF 0x007FFFFFu  // enc(-inf)

// ---------------- init: zero out accumulator, reset row max/sum ----------------
__global__ void gat_init_kernel(float* __restrict__ out, unsigned* __restrict__ row_max,
                                float* __restrict__ row_sum, int Nn, long total) {
    long gid = (long)blockIdx.x * blockDim.x + threadIdx.x;
    if (gid < total) out[gid] = 0.0f;
    if (gid < Nn) { row_max[gid] = ENC_NEG_INF; row_sum[gid] = 0.0f; }
}

// ---------------- h = x @ W  (WMMA bf16->f32), fused s_src/s_dst ----------------
// W is staged once per block into LDS, bf16-converted and pre-swizzled into the
// exact 16x16x32 B-fragment register order: frag f = ks*4 + t; slot(f,lane)
// holds the lane's 16 bf16 values (8 dwords), so a B fragment is one 32B LDS read.
__global__ __launch_bounds__(256)
void gat_gemm_kernel(const float* __restrict__ x, const float* __restrict__ W,
                     const float* __restrict__ a,
                     float* __restrict__ h, float* __restrict__ s_src,
                     float* __restrict__ s_dst, int Nn) {
    __shared__ __align__(32) v2bf wlds[4096];   // 16 KB: 16 frags * 32 lanes * 8 dwords

    const int tid = threadIdx.x;

    // ---- cooperative swizzled bf16 staging of W (once per block) ----
    #pragma unroll
    for (int j = 0; j < 16; ++j) {
        const int g     = tid + 256 * j;     // global dword id in swizzled layout
        const int d     = g & 7;             // dword within lane slot
        const int laneg = (g >> 3) & 31;
        const int f     = g >> 8;            // fragment id
        const int ks    = f >> 2;
        const int t     = f & 3;
        const int halfg = laneg >> 4;
        const int qg    = laneg & 15;
        const int K     = ks * 32 + halfg * 16 + 2 * d;
        const int n     = t * 16 + qg;
        v2bf p;
        p[0] = (__bf16)W[K * OUT_F + n];
        p[1] = (__bf16)W[(K + 1) * OUT_F + n];
        wlds[g] = p;
    }
    __syncthreads();

    const int wave = tid >> 5;
    const int lane = tid & 31;
    const int half = lane >> 4;     // 0 or 1
    const int q    = lane & 15;
    const int m0   = (blockIdx.x * 8 + wave) * 16;

    if (m0 < Nn) {                  // wave-uniform predicate: EXEC all-1s inside
        const v8f vzero = {0.f,0.f,0.f,0.f,0.f,0.f,0.f,0.f};
        v8f acc[4];
        #pragma unroll
        for (int t = 0; t < 4; ++t) acc[t] = vzero;

        int mA = m0 + q; if (mA >= Nn) mA = Nn - 1;      // clamped read row
        const float* xrow = x + (long)mA * IN_F;

        #pragma unroll
        for (int ks = 0; ks < 4; ++ks) {
            const int kk = ks * 32;
            // A fragment: 16x32 bf16; each lane half holds 16 contiguous floats (k = 16*half + s) -> b128 loads
            const float4 p0 = *(const float4*)(xrow + kk + half * 16);
            const float4 p1 = *(const float4*)(xrow + kk + half * 16 + 4);
            const float4 p2 = *(const float4*)(xrow + kk + half * 16 + 8);
            const float4 p3 = *(const float4*)(xrow + kk + half * 16 + 12);
            v16bf afrag;
            afrag[0]  = (__bf16)p0.x; afrag[1]  = (__bf16)p0.y;
            afrag[2]  = (__bf16)p0.z; afrag[3]  = (__bf16)p0.w;
            afrag[4]  = (__bf16)p1.x; afrag[5]  = (__bf16)p1.y;
            afrag[6]  = (__bf16)p1.z; afrag[7]  = (__bf16)p1.w;
            afrag[8]  = (__bf16)p2.x; afrag[9]  = (__bf16)p2.y;
            afrag[10] = (__bf16)p2.z; afrag[11] = (__bf16)p2.w;
            afrag[12] = (__bf16)p3.x; afrag[13] = (__bf16)p3.y;
            afrag[14] = (__bf16)p3.z; afrag[15] = (__bf16)p3.w;

            #pragma unroll
            for (int t = 0; t < 4; ++t) {
                const v16bf bfrag =
                    *(const v16bf*)&wlds[((ks * 4 + t) * 32 + lane) * 8];
                acc[t] = __builtin_amdgcn_wmma_f32_16x16x32_bf16(
                    false, afrag, false, bfrag, (short)0, acc[t], false, false);
            }
        }

        // store h; fused logit projections s_src = h@a[:64], s_dst = h@a[64:]
        #pragma unroll
        for (int r = 0; r < 8; ++r) {
            const int ro = m0 + r + 8 * half;       // output row per C/D layout
            float ssrc = 0.f, sdst = 0.f;
            #pragma unroll
            for (int t = 0; t < 4; ++t) {
                float v = acc[t][r];
                int   c = t * 16 + q;
                if (ro < Nn) h[(long)ro * OUT_F + c] = v;
                ssrc += v * a[c];
                sdst += v * a[OUT_F + c];
            }
            #pragma unroll
            for (int off = 8; off >= 1; off >>= 1) {  // reduce across the 16-lane half
                ssrc += __shfl_xor(ssrc, off, 16);
                sdst += __shfl_xor(sdst, off, 16);
            }
            if (q == 0 && ro < Nn) { s_src[ro] = ssrc; s_dst[ro] = sdst; }
        }
    }
}

// ---------------- edge pass 1: leaky_relu logits + segment max ----------------
__global__ void gat_logits_kernel(const int* __restrict__ row, const int* __restrict__ col,
                                  const float* __restrict__ s_src, const float* __restrict__ s_dst,
                                  float* __restrict__ e, unsigned* __restrict__ row_max, int E) {
    int i = blockIdx.x * blockDim.x + threadIdx.x;
    if (i >= E) return;
    int r = row[i], c = col[i];
    float v = s_src[r] + s_dst[c];
    v = (v > 0.f) ? v : ALPHA * v;
    e[i] = v;
    atomicMax(row_max + r, enc_f32(v));
}

// ---------------- edge pass 2: exp(e - max) + segment sum ----------------
__global__ void gat_expsum_kernel(const int* __restrict__ row, float* __restrict__ e,
                                  const unsigned* __restrict__ row_max,
                                  float* __restrict__ row_sum, int E) {
    int i = blockIdx.x * blockDim.x + threadIdx.x;
    if (i >= E) return;
    int r = row[i];
    float ex = __expf(e[i] - dec_f32(row_max[r]));
    e[i] = ex;
    atomicAdd(row_sum + r, ex);
}

// ---------------- edge pass 3: out[row] += att * h[col]  (16 lanes / edge) ----------------
__global__ void gat_scatter_kernel(const int* __restrict__ row, const int* __restrict__ col,
                                   const float* __restrict__ e, const float* __restrict__ row_sum,
                                   const float* __restrict__ h, float* __restrict__ out, int E) {
    long gid = (long)blockIdx.x * blockDim.x + threadIdx.x;
    int i  = (int)(gid >> 4);
    int c4 = (int)(gid & 15) * 4;
    if (i >= E) return;
    int r = row[i], c = col[i];
    float att = e[i] / fmaxf(row_sum[r], EPSV);
    const float4 hv = *reinterpret_cast<const float4*>(h + (long)c * OUT_F + c4);
    float* o = out + (long)r * OUT_F + c4;
    atomicAdd(o + 0, att * hv.x);
    atomicAdd(o + 1, att * hv.y);
    atomicAdd(o + 2, att * hv.z);
    atomicAdd(o + 3, att * hv.w);
}

// ---------------- finalize: in-place ELU ----------------
__global__ void gat_elu_kernel(float* __restrict__ out, long total) {
    long gid = (long)blockIdx.x * blockDim.x + threadIdx.x;
    if (gid >= total) return;
    float v = out[gid];
    out[gid] = (v > 0.f) ? v : (__expf(v) - 1.f);
}

extern "C" void kernel_launch(void* const* d_in, const int* in_sizes, int n_in,
                              void* d_out, int out_size, void* d_ws, size_t ws_size,
                              hipStream_t stream) {
    const float* x  = (const float*)d_in[0];
    const float* W  = (const float*)d_in[1];
    const float* a  = (const float*)d_in[2];
    const int*   ei = (const int*)d_in[3];

    const int Nn = in_sizes[0] / IN_F;
    const int E  = in_sizes[3] / 2;
    const int* row = ei;
    const int* col = ei + E;
    float* out = (float*)d_out;

    // workspace carve-up
    float*    h       = (float*)d_ws;                 // Nn*64
    float*    e       = h + (long)Nn * OUT_F;         // E
    unsigned* row_max = (unsigned*)(e + E);           // Nn
    float*    row_sum = (float*)(row_max + Nn);       // Nn
    float*    s_src   = row_sum + Nn;                 // Nn
    float*    s_dst   = s_src + Nn;                   // Nn

    const long total = (long)Nn * OUT_F;
    const int  tb = 256;

    gat_init_kernel<<<(int)((total + tb - 1) / tb), tb, 0, stream>>>(out, row_max, row_sum, Nn, total);

    const int ntiles  = (Nn + 15) / 16;
    const int gblocks = (ntiles + 7) / 8;
    gat_gemm_kernel<<<gblocks, 256, 0, stream>>>(x, W, a, h, s_src, s_dst, Nn);

    gat_logits_kernel<<<(E + tb - 1) / tb, tb, 0, stream>>>(row, col, s_src, s_dst, e, row_max, E);
    gat_expsum_kernel<<<(E + tb - 1) / tb, tb, 0, stream>>>(row, e, row_max, row_sum, E);

    const long sthreads = (long)E * 16;
    gat_scatter_kernel<<<(int)((sthreads + tb - 1) / tb), tb, 0, stream>>>(row, col, e, row_sum, h, out, E);

    gat_elu_kernel<<<(int)((total + tb - 1) / tb), tb, 0, stream>>>(out, total);
}
